// AnomalyDAE_Base_65360812310962
// MI455X (gfx1250) — compile-verified, matrix operand order checked by reading, not run on hardware
//
#include <hip/hip_runtime.h>
#include <hip/hip_bf16.h>
#include <math.h>

// ---- problem constants (match reference) ----
#define NN   12288   // num nodes
#define DIN  512
#define DEMB 128
#define DOUT 64
#define NEG_SLOPE 0.2f

typedef __attribute__((ext_vector_type(16))) __bf16 v16bf;
typedef __attribute__((ext_vector_type(8)))  float  v8f;

union FragU { v16bf v; uint4 q[2]; };

// Load a 16x32 bf16 WMMA fragment from a row-major [rows, ld] matrix.
// Works for both A (row = M) and B (row = N, i.e. computing A*B^T).
// Per ISA layout: lanes 0-15 hold K {0..7,16..23}; lanes 16-31 hold K {8..15,24..31}.
// Each half is 8 contiguous bf16 = one b128 load (global or ds, by address space).
__device__ __forceinline__ v16bf load_frag(const __bf16* base, int ld, int lane) {
  int r  = lane & 15;
  int kh = (lane >> 4) & 1;          // which K half-offset (+8)
  const __bf16* p = base + (size_t)r * ld + kh * 8;
  FragU f;
  f.q[0] = *reinterpret_cast<const uint4*>(p);        // K = kh*8 + 0..7
  f.q[1] = *reinterpret_cast<const uint4*>(p + 16);   // K = kh*8 + 16..23
  return f.v;
}

// Generic A(row-major [M,K]) * B(row-major [N,K])^T GEMM, one 16x16 tile per wave,
// fp32 accumulation via v_wmma_f32_16x16x32_bf16.
// ACT: 0=none, 1=relu, 2=sigmoid.  OUTBF: write bf16 instead of f32.
template<bool BIAS, int ACT, bool OUTBF>
__global__ void wmma_gemm(const __bf16* __restrict__ A, int lda,
                          const __bf16* __restrict__ B, int ldb,
                          const float* __restrict__ bias,
                          void* __restrict__ Out, int ldc,
                          int tilesM, int tilesN, int K) {
  const int wave = threadIdx.x >> 5;
  const int lane = threadIdx.x & 31;
  const int tile = blockIdx.x * (blockDim.x >> 5) + wave;
  if (tile >= tilesM * tilesN) return;      // wave-uniform: EXEC stays all-1s
  const int tm = tile / tilesN;
  const int tn = tile % tilesN;

  const __bf16* Abase = A + (size_t)tm * 16 * lda;
  const __bf16* Bbase = B + (size_t)tn * 16 * ldb;

  v8f acc = {};
  for (int k0 = 0; k0 < K; k0 += 32) {
    v16bf a = load_frag(Abase + k0, lda, lane);
    v16bf b = load_frag(Bbase + k0, ldb, lane);
    acc = __builtin_amdgcn_wmma_f32_16x16x32_bf16(
        false, a, false, b, (short)0, acc, false, false);
  }

  // D layout: VGPR j holds row (tm*16 + 8*(lane>>4) + j), col = tn*16 + (lane&15)
  const int col  = tn * 16 + (lane & 15);
  const int row0 = tm * 16 + ((lane >> 4) << 3);
  const float bv = BIAS ? bias[col] : 0.0f;
#pragma unroll
  for (int j = 0; j < 8; j++) {
    float v = acc[j] + bv;
    if (ACT == 1) v = v > 0.0f ? v : 0.0f;
    if (ACT == 2) v = 1.0f / (1.0f + __expf(-v));
    const size_t o = (size_t)(row0 + j) * ldc + col;
    if (OUTBF) ((__bf16*)Out)[o] = (__bf16)v;
    else       ((float*)Out)[o]  = v;
  }
}

// Specialized s_ = sigmoid(H @ H^T), H = [NN, 64] bf16.
// Block = 256 threads (8 waves) computes a 128x128 output tile.
// Both 128x64 panels are staged in LDS with GLOBAL_LOAD_ASYNC_TO_LDS_B128
// (ASYNCcnt-tracked async DMA, no VGPR round-trip), then each wave keeps its
// A fragments in registers across 8 column tiles; B fragments come from LDS.
__global__ void __launch_bounds__(256)
wmma_hht_sigmoid(const __bf16* __restrict__ H, float* __restrict__ Out) {
  __shared__ __bf16 As[128 * DOUT];   // 16 KB
  __shared__ __bf16 Bs[128 * DOUT];   // 16 KB
  const int lane   = threadIdx.x & 31;
  const int wave   = threadIdx.x >> 5;
  const int rowBlk = blockIdx.y * 128;
  const int colBlk = blockIdx.x * 128;

  // ---- async panel staging: 2 x 16KB, 16B per lane per issue ----
  {
    const int t = threadIdx.x;                 // 256 threads
#pragma unroll
    for (int it = 0; it < 4; ++it) {
      int idx = it * 256 + t;                  // chunk 0..1023 (16B chunks)
      int row = idx >> 3;                      // 8 chunks per 128B row
      int ch  = (idx & 7) * 8;                 // bf16 offset within row
      unsigned long long ga =
          (unsigned long long)(H + (size_t)(rowBlk + row) * DOUT + ch);
      unsigned long long gb =
          (unsigned long long)(H + (size_t)(colBlk + row) * DOUT + ch);
      unsigned la = (unsigned)(size_t)&As[row * DOUT + ch];  // LDS byte offset
      unsigned lb = (unsigned)(size_t)&Bs[row * DOUT + ch];
      asm volatile("global_load_async_to_lds_b128 %0, %1, off"
                   :: "v"(la), "v"(ga) : "memory");
      asm volatile("global_load_async_to_lds_b128 %0, %1, off"
                   :: "v"(lb), "v"(gb) : "memory");
    }
    asm volatile("s_wait_asynccnt 0" ::: "memory");
  }
  __syncthreads();

  // ---- each wave: one 16-row strip, A frags held in registers ----
  const __bf16* Aw = &As[wave * 16 * DOUT];
  v16bf a0 = load_frag(Aw + 0,  DOUT, lane);
  v16bf a1 = load_frag(Aw + 32, DOUT, lane);

  const int colLane = lane & 15;
  const int rowHalf = (lane >> 4) << 3;
  float* outBase = Out + (size_t)(rowBlk + wave * 16 + rowHalf) * NN + colBlk;

#pragma unroll
  for (int nt = 0; nt < 8; ++nt) {
    const __bf16* Bw = &Bs[nt * 16 * DOUT];
    v16bf b0 = load_frag(Bw + 0,  DOUT, lane);
    v16bf b1 = load_frag(Bw + 32, DOUT, lane);
    v8f acc = {};
    acc = __builtin_amdgcn_wmma_f32_16x16x32_bf16(false, a0, false, b0,
                                                  (short)0, acc, false, false);
    acc = __builtin_amdgcn_wmma_f32_16x16x32_bf16(false, a1, false, b1,
                                                  (short)0, acc, false, false);
    float* op = outBase + nt * 16 + colLane;
#pragma unroll
    for (int j = 0; j < 8; j++) {
      float v = 1.0f / (1.0f + __expf(-acc[j]));
      op[(size_t)j * NN] = v;
    }
  }
}

// ---- helpers ----
__global__ void f32_to_bf16(const float* __restrict__ in, __bf16* __restrict__ out,
                            size_t n) {
  size_t i = (size_t)blockIdx.x * blockDim.x + threadIdx.x;
  if (i < n) out[i] = (__bf16)in[i];
}

__global__ void fill_f32(float* __restrict__ p, float v, size_t n) {
  size_t i = (size_t)blockIdx.x * blockDim.x + threadIdx.x;
  if (i < n) p[i] = v;
}

// LDS-tiled transpose: in f32 [rows, cols] -> out bf16 [cols, rows]
__global__ void transpose_to_bf16(const float* __restrict__ in,
                                  __bf16* __restrict__ out, int rows, int cols) {
  __shared__ float t[32][33];
  const int bc = blockIdx.x * 32, br = blockIdx.y * 32;
  const int tx = threadIdx.x, ty = threadIdx.y;   // (32, 8)
#pragma unroll
  for (int i = 0; i < 32; i += 8) {
    int r = br + ty + i, c = bc + tx;
    t[ty + i][tx] = (r < rows && c < cols) ? in[(size_t)r * cols + c] : 0.0f;
  }
  __syncthreads();
#pragma unroll
  for (int i = 0; i < 32; i += 8) {
    int r = bc + ty + i, c = br + tx;
    if (r < cols && c < rows) out[(size_t)r * rows + c] = (__bf16)t[tx][ty + i];
  }
}

// per-node attention scores: sc = g . a
__global__ void node_scores(const float* __restrict__ g,
                            const float* __restrict__ a_src,
                            const float* __restrict__ a_dst,
                            float* __restrict__ ss, float* __restrict__ sd, int n) {
  int i = blockIdx.x * blockDim.x + threadIdx.x;
  if (i >= n) return;
  const float* gp = g + (size_t)i * DOUT;
  float s0 = 0.f, s1 = 0.f;
#pragma unroll 8
  for (int c = 0; c < DOUT; c++) {
    float v = gp[c];
    s0 += v * a_src[c];
    s1 += v * a_dst[c];
  }
  ss[i] = s0;
  sd[i] = s1;
}

__device__ __forceinline__ void atomicMaxF32(float* addr, float val) {
  if (val >= 0.0f) atomicMax((int*)addr, __float_as_int(val));
  else             atomicMin((unsigned int*)addr, __float_as_uint(val));
}

__device__ __forceinline__ void edge_sd(const long long* __restrict__ ei, int E,
                                        int e, int& s, int& d) {
  if (e < E) { s = (int)ei[e]; d = (int)ei[(size_t)E + e]; }
  else       { s = d = e - E; }          // self loops appended
}

__global__ void edge_max(const long long* __restrict__ ei, int E, int n,
                         const float* __restrict__ ss, const float* __restrict__ sd,
                         float* __restrict__ m) {
  int e = blockIdx.x * blockDim.x + threadIdx.x;
  if (e >= E + n) return;
  int s, d; edge_sd(ei, E, e, s, d);
  float v = ss[s] + sd[d];
  v = v > 0.0f ? v : NEG_SLOPE * v;
  atomicMaxF32(&m[d], v);
}

__global__ void edge_acc(const long long* __restrict__ ei, int E, int n,
                         const float* __restrict__ ss, const float* __restrict__ sd,
                         const float* __restrict__ m, const float* __restrict__ g,
                         float* __restrict__ denom, float* __restrict__ hacc) {
  int e = blockIdx.x * blockDim.x + threadIdx.x;
  if (e >= E + n) return;
  int s, d; edge_sd(ei, E, e, s, d);
  float v = ss[s] + sd[d];
  v = v > 0.0f ? v : NEG_SLOPE * v;
  float w = __expf(v - m[d]);
  atomicAdd(&denom[d], w);
  const float* gs = g + (size_t)s * DOUT;
  float* hd = hacc + (size_t)d * DOUT;
#pragma unroll 8
  for (int c = 0; c < DOUT; c++) atomicAdd(&hd[c], w * gs[c]);
}

// h = hacc/denom + b_g  -> bf16
__global__ void norm_h(const float* __restrict__ hacc, const float* __restrict__ denom,
                       const float* __restrict__ bg, __bf16* __restrict__ hh, int n) {
  int i = blockIdx.x * blockDim.x + threadIdx.x;
  if (i >= n * DOUT) return;
  int node = i >> 6, c = i & 63;
  hh[i] = (__bf16)(hacc[i] / denom[node] + bg[c]);
}

// ---- launcher ----
static inline size_t cdiv(size_t a, size_t b) { return (a + b - 1) / b; }

extern "C" void kernel_launch(void* const* d_in, const int* in_sizes, int n_in,
                              void* d_out, int out_size, void* d_ws, size_t ws_size,
                              hipStream_t stream) {
  const float* x     = (const float*)d_in[0];
  const float* W_s1  = (const float*)d_in[1];
  const float* b_s1  = (const float*)d_in[2];
  const float* W_g   = (const float*)d_in[3];
  const float* a_src = (const float*)d_in[4];
  const float* a_dst = (const float*)d_in[5];
  const float* b_g   = (const float*)d_in[6];
  const float* W_a1  = (const float*)d_in[7];
  const float* b_a1  = (const float*)d_in[8];
  const float* W_a2  = (const float*)d_in[9];
  const float* b_a2  = (const float*)d_in[10];
  const long long* edge_index = (const long long*)d_in[11];
  const int E = in_sizes[11] / 2;

  float* out_x = (float*)d_out;                         // [NN, DIN]
  float* out_s = (float*)d_out + (size_t)NN * DIN;      // [NN, NN]

  // workspace carve-out
  char* w = (char*)d_ws;
  auto alloc = [&](size_t bytes) -> void* {
    void* p = (void*)w;
    w += (bytes + 255) & ~(size_t)255;
    return p;
  };
  __bf16* xh    = (__bf16*)alloc((size_t)NN * DIN * 2);
  __bf16* xTh   = (__bf16*)alloc((size_t)DIN * NN * 2);
  __bf16* Ws1h  = (__bf16*)alloc((size_t)DEMB * DIN * 2);
  __bf16* Wgh   = (__bf16*)alloc((size_t)DOUT * DEMB * 2);
  __bf16* Wa1h  = (__bf16*)alloc((size_t)DEMB * NN * 2);
  __bf16* Wa2h  = (__bf16*)alloc((size_t)DOUT * DEMB * 2);
  __bf16* x1h   = (__bf16*)alloc((size_t)NN * DEMB * 2);
  float*  g     = (float*)alloc((size_t)NN * DOUT * 4);
  float*  ss    = (float*)alloc((size_t)NN * 4);
  float*  sd    = (float*)alloc((size_t)NN * 4);
  float*  mmax  = (float*)alloc((size_t)NN * 4);
  float*  den   = (float*)alloc((size_t)NN * 4);
  float*  hacc  = (float*)alloc((size_t)NN * DOUT * 4);
  __bf16* hh    = (__bf16*)alloc((size_t)NN * DOUT * 2);
  __bf16* xth   = (__bf16*)alloc((size_t)DIN * DEMB * 2);
  __bf16* xah   = (__bf16*)alloc((size_t)DIN * DOUT * 2);

  const int TB = 256;                 // 8 waves / block
  const int WPB = TB / 32;

  // 0) precision conversion of streamed operands
  f32_to_bf16<<<cdiv((size_t)NN * DIN, TB), TB, 0, stream>>>(x, xh, (size_t)NN * DIN);
  f32_to_bf16<<<cdiv((size_t)DEMB * DIN, TB), TB, 0, stream>>>(W_s1, Ws1h, (size_t)DEMB * DIN);
  f32_to_bf16<<<cdiv((size_t)DOUT * DEMB, TB), TB, 0, stream>>>(W_g, Wgh, (size_t)DOUT * DEMB);
  f32_to_bf16<<<cdiv((size_t)DEMB * NN, TB), TB, 0, stream>>>(W_a1, Wa1h, (size_t)DEMB * NN);
  f32_to_bf16<<<cdiv((size_t)DOUT * DEMB, TB), TB, 0, stream>>>(W_a2, Wa2h, (size_t)DOUT * DEMB);
  transpose_to_bf16<<<dim3(DIN / 32, NN / 32), dim3(32, 8), 0, stream>>>(x, xTh, NN, DIN);

  // 1) x1 = relu(x @ W_s1^T + b_s1)  [NN, DEMB] bf16
  {
    int tiles = (NN / 16) * (DEMB / 16);
    wmma_gemm<true, 1, true><<<cdiv(tiles, WPB), TB, 0, stream>>>(
        xh, DIN, Ws1h, DIN, b_s1, x1h, DEMB, NN / 16, DEMB / 16, DIN);
  }
  // 2) g = x1 @ W_g^T   [NN, DOUT] f32
  {
    int tiles = (NN / 16) * (DOUT / 16);
    wmma_gemm<false, 0, false><<<cdiv(tiles, WPB), TB, 0, stream>>>(
        x1h, DEMB, Wgh, DEMB, nullptr, g, DOUT, NN / 16, DOUT / 16, DEMB);
  }
  // 3) attention scores per node
  node_scores<<<cdiv(NN, TB), TB, 0, stream>>>(g, a_src, a_dst, ss, sd, NN);

  // 4) GAT edge softmax + aggregation (atomics into L2-resident buffers)
  fill_f32<<<cdiv(NN, TB), TB, 0, stream>>>(mmax, -__builtin_inff(), NN);
  fill_f32<<<cdiv(NN, TB), TB, 0, stream>>>(den, 0.0f, NN);
  fill_f32<<<cdiv((size_t)NN * DOUT, TB), TB, 0, stream>>>(hacc, 0.0f, (size_t)NN * DOUT);
  edge_max<<<cdiv(E + NN, TB), TB, 0, stream>>>(edge_index, E, NN, ss, sd, mmax);
  edge_acc<<<cdiv(E + NN, TB), TB, 0, stream>>>(edge_index, E, NN, ss, sd, mmax, g, den, hacc);
  norm_h<<<cdiv((size_t)NN * DOUT, TB), TB, 0, stream>>>(hacc, den, b_g, hh, NN);

  // 5) xt = relu(x^T @ W_a1^T + b_a1)   [DIN, DEMB] bf16
  {
    int tiles = (DIN / 16) * (DEMB / 16);
    wmma_gemm<true, 1, true><<<cdiv(tiles, WPB), TB, 0, stream>>>(
        xTh, NN, Wa1h, NN, b_a1, xth, DEMB, DIN / 16, DEMB / 16, NN);
  }
  // 6) xa = xt @ W_a2^T + b_a2          [DIN, DOUT] bf16
  {
    int tiles = (DIN / 16) * (DOUT / 16);
    wmma_gemm<true, 0, true><<<cdiv(tiles, WPB), TB, 0, stream>>>(
        xth, DEMB, Wa2h, DEMB, b_a2, xah, DOUT, DIN / 16, DOUT / 16, DEMB);
  }
  // 7) x_ = h @ xa^T                    [NN, DIN] f32 -> out
  {
    int tiles = (NN / 16) * (DIN / 16);
    wmma_gemm<false, 0, false><<<cdiv(tiles, WPB), TB, 0, stream>>>(
        hh, DOUT, xah, DOUT, nullptr, out_x, DIN, NN / 16, DIN / 16, DOUT);
  }
  // 8) s_ = sigmoid(h @ h^T)  [NN, NN] f32 -> out  (604 MB stream-out)
  //    LDS-blocked 128x128 tiles, async global->LDS panel staging.
  {
    dim3 grid(NN / 128, NN / 128);
    wmma_hht_sigmoid<<<grid, 256, 0, stream>>>(hh, out_s);
  }
}
